// Adapter_69647189672531
// MI455X (gfx1250) — compile-verified, matrix operand order checked by reading, not run on hardware
//
#include <hip/hip_runtime.h>
#include <hip/hip_bf16.h>
#include <math.h>

// ---------------------------------------------------------------------------
// CDNA5 (gfx1250, wave32) WMMA implementation of the adapter block.
// All heavy einsum/conv work runs through v_wmma_f32_16x16x32_bf16.
// ---------------------------------------------------------------------------

typedef __attribute__((ext_vector_type(16))) __bf16 v16bf;
typedef __attribute__((ext_vector_type(8)))  float  v8f;

#define BMT 128
#define BNT 128
#define BKT 32
#define LDP 40  // padded LDS row stride (bf16 elems); 80B rows keep 16B align

struct GP {
  const float* A; const float* Bw; float* C;
  const float* bias; const float* scale; const float* shift; const float* Res;
  long long aB, aR, aC;     // A: batch, row(m), col(k) strides
  long long bB, bR, bC;     // B: batch, row(n), col(k) strides
  long long cB, cR, cC;     // C strides
  long long rB, rR, rC;     // residual strides
  int M, N, K;
  int aMode;                // 0 affine, 1 implicit 3x3 conv (m=spatial, k=q*Cc+c)
  int bMode;                // 0 affine, 1 conv weight [n][c][3][3]
  int Cc, Wc, Hc;
  int act;                  // 0 none, 1 relu, 2 gelu(exact), 3 sigmoid
  float alpha;
};

__device__ __forceinline__ unsigned short f2bfu(float f) {
  union { float f; unsigned u; } a; a.f = f;
  return (unsigned short)((a.u + 0x7FFFu + ((a.u >> 16) & 1u)) >> 16);
}
__device__ __forceinline__ __bf16 f2bf(float f) {
  unsigned short h = f2bfu(f);
  __bf16 o; __builtin_memcpy(&o, &h, 2); return o;
}

// --- generic scalar loaders (fallback path only) ---------------------------
__device__ __forceinline__ float loadA(const GP& p, int z, int m, int k) {
  if (m >= p.M || k >= p.K) return 0.f;
  if (p.aMode == 0)
    return p.A[(long long)z * p.aB + (long long)m * p.aR + (long long)k * p.aC];
  int q = k / p.Cc, c = k - q * p.Cc;      // k = q*Cc + c, q = ky*3+kx
  int dy = q / 3 - 1, dx = q - (q / 3) * 3 - 1;
  int y = m / p.Wc, x = m - y * p.Wc;
  int xx = x + dx, yy = y + dy;
  if (xx < 0 || xx >= p.Wc || yy < 0 || yy >= p.Hc) return 0.f;
  return p.A[(long long)z * p.aB + (long long)c * p.aC +
             (long long)(yy * p.Wc + xx) * p.aR];
}
__device__ __forceinline__ float loadB(const GP& p, int z, int n, int k) {
  if (n >= p.N || k >= p.K) return 0.f;
  if (p.bMode == 0)
    return p.Bw[(long long)z * p.bB + (long long)n * p.bR + (long long)k * p.bC];
  int q = k / p.Cc, c = k - q * p.Cc;
  return p.Bw[(long long)n * p.bR + (long long)c * 9 + q];
}

// --- vectorized loaders (k-contiguous, 16B-aligned) ------------------------
// conv-mode uniform values (cT,dy,dx) are hoisted to once-per-k-tile; the
// per-thread spatial rows (y,x) are hoisted to once-per-kernel.
__device__ __forceinline__ float4 load4tail(const float* ptr, int k, int K) {
  float4 r = make_float4(0.f, 0.f, 0.f, 0.f);
  if (k + 3 < K) r = *(const float4*)ptr;
  else for (int i = 0; i < 4; ++i) if (k + i < K) (&r.x)[i] = ptr[i];
  return r;
}
__device__ __forceinline__ float4 load4A_aff(const GP& p, int z, int m, int k) {
  if (m >= p.M) return make_float4(0.f, 0.f, 0.f, 0.f);
  long long base = (long long)z * p.aB + (long long)m * p.aR + k;
  if (k + BKT + 3 < p.K)
    __builtin_prefetch((const void*)(p.A + base + BKT), 0, 3);
  return load4tail(p.A + base, k, p.K);
}
__device__ __forceinline__ float4 load4A_conv(const GP& p, int z, int m,
                                              int y, int x, int c,
                                              int dy, int dx) {
  float4 r = make_float4(0.f, 0.f, 0.f, 0.f);
  if (m >= p.M) return r;
  int xx = x + dx, yy = y + dy;
  if (xx < 0 || xx >= p.Wc || yy < 0 || yy >= p.Hc) return r;
  long long base = (long long)z * p.aB + (long long)(yy * p.Wc + xx) * p.aR + c;
  return *(const float4*)(p.A + base);   // conv K is a multiple of 4
}
__device__ __forceinline__ float4 load4B_aff(const GP& p, int z, int n, int k) {
  if (n >= p.N) return make_float4(0.f, 0.f, 0.f, 0.f);
  long long base = (long long)z * p.bB + (long long)n * p.bR + k;
  if (k + BKT + 3 < p.K)
    __builtin_prefetch((const void*)(p.Bw + base + BKT), 0, 3);
  return load4tail(p.Bw + base, k, p.K);
}
__device__ __forceinline__ void store4bf(__bf16* dst, float4 v) {
  uint2 u;
  u.x = (unsigned)f2bfu(v.x) | ((unsigned)f2bfu(v.y) << 16);
  u.y = (unsigned)f2bfu(v.z) | ((unsigned)f2bfu(v.w) << 16);
  *(uint2*)dst = u;
}

template <int FA, int FB>
__global__ __launch_bounds__(256) void wmma_gemm(GP p) {
  __shared__ __bf16 As[BMT][LDP];
  __shared__ __bf16 Bs[BNT][LDP];
  const int tid  = threadIdx.x;
  const int lane = tid & 31;
  const int wave = tid >> 5;
  const int wm = wave >> 1;       // 0..3  -> 32-row strip
  const int wn = wave & 1;        // 0..1  -> 64-col strip
  const int z  = blockIdx.z;
  const int bm = blockIdx.x * BMT;
  const int bn = blockIdx.y * BNT;

  // loop-invariant fill coordinates (fast paths: 4 float4 per thread)
  int amm[4], akk[4], aax[4], aay[4];
  if (FA) {
#pragma unroll
    for (int it = 0; it < 4; ++it) {
      int e = it * 256 + tid;
      amm[it] = e >> 3;
      akk[it] = (e & 7) << 2;
      if (p.aMode == 1) {
        int m = bm + amm[it];
        int y = m / p.Wc;
        aay[it] = y; aax[it] = m - y * p.Wc;
      } else { aay[it] = 0; aax[it] = 0; }
    }
  }

  v8f acc[2][4];
#pragma unroll
  for (int i = 0; i < 2; ++i)
#pragma unroll
    for (int j = 0; j < 4; ++j)
#pragma unroll
      for (int r = 0; r < 8; ++r) acc[i][j][r] = 0.f;

  const int mrow = lane & 15;
  const int kb   = (lane >> 4) << 3;   // 0 (lanes 0-15) / 8 (lanes 16-31)

  for (int kt = 0; kt < p.K; kt += BKT) {
    if (FA) {
      if (p.aMode == 1) {
        // k-tile-uniform conv tap (tile never crosses q: Cc % BKT == 0)
        int qT = kt / p.Cc;
        int dyT = qT / 3 - 1, dxT = qT - (qT / 3) * 3 - 1;
        int cT = kt - qT * p.Cc;
#pragma unroll
        for (int it = 0; it < 4; ++it)
          store4bf(&As[amm[it]][akk[it]],
                   load4A_conv(p, z, bm + amm[it], aay[it], aax[it],
                               cT + akk[it], dyT, dxT));
      } else {
#pragma unroll
        for (int it = 0; it < 4; ++it)
          store4bf(&As[amm[it]][akk[it]],
                   load4A_aff(p, z, bm + amm[it], kt + akk[it]));
      }
    } else {
#pragma unroll
      for (int it = 0; it < 16; ++it) {
        int e = it * 256 + tid;
        int mm = e >> 5, kk = e & 31;
        As[mm][kk] = f2bf(loadA(p, z, bm + mm, kt + kk));
      }
    }
    if (FB) {
#pragma unroll
      for (int it = 0; it < 4; ++it) {
        int e = it * 256 + tid;
        int mm = e >> 3, kk = (e & 7) << 2;
        store4bf(&Bs[mm][kk], load4B_aff(p, z, bn + mm, kt + kk));
      }
    } else {
#pragma unroll
      for (int it = 0; it < 16; ++it) {
        int e = it * 256 + tid;
        int mm = e >> 5, kk = e & 31;
        Bs[mm][kk] = f2bf(loadB(p, z, bn + mm, kt + kk));
      }
    }
    __syncthreads();

    // 16-bit A 16x32 fragment layout (ISA 7.12.2): lane 0-15 M=lane, K pairs
    // {kb+2v, kb+2v+1} in regs 0-3, {16+kb+2v,...} in regs 4-7.
    v16bf af[2], bfm[4];
#pragma unroll
    for (int i = 0; i < 2; ++i) {
      int m = wm * 32 + i * 16 + mrow;
#pragma unroll
      for (int v = 0; v < 4; ++v) {
        af[i][2 * v]         = As[m][kb + 2 * v];
        af[i][2 * v + 1]     = As[m][kb + 2 * v + 1];
        af[i][8 + 2 * v]     = As[m][16 + kb + 2 * v];
        af[i][8 + 2 * v + 1] = As[m][16 + kb + 2 * v + 1];
      }
    }
#pragma unroll
    for (int j = 0; j < 4; ++j) {
      int n = wn * 64 + j * 16 + mrow;
#pragma unroll
      for (int v = 0; v < 4; ++v) {
        bfm[j][2 * v]         = Bs[n][kb + 2 * v];
        bfm[j][2 * v + 1]     = Bs[n][kb + 2 * v + 1];
        bfm[j][8 + 2 * v]     = Bs[n][16 + kb + 2 * v];
        bfm[j][8 + 2 * v + 1] = Bs[n][16 + kb + 2 * v + 1];
      }
    }
#pragma unroll
    for (int i = 0; i < 2; ++i)
#pragma unroll
      for (int j = 0; j < 4; ++j)
        acc[i][j] = __builtin_amdgcn_wmma_f32_16x16x32_bf16(
            false, af[i], false, bfm[j], (short)0, acc[i][j], false, false);
    __syncthreads();
  }

  // Epilogue. C/D layout: reg r, lanes 0-15 -> M=r, lanes 16-31 -> M=r+8, N=lane&15.
  const int mh = (lane >> 4) << 3;
#pragma unroll
  for (int i = 0; i < 2; ++i)
#pragma unroll
    for (int j = 0; j < 4; ++j) {
      int n = bn + wn * 64 + j * 16 + mrow;
      if (n >= p.N) continue;
#pragma unroll
      for (int r = 0; r < 8; ++r) {
        int m = bm + wm * 32 + i * 16 + mh + r;
        if (m >= p.M) continue;
        float c = acc[i][j][r] * p.alpha;
        if (p.bias)  c += p.bias[n];
        if (p.scale) c = c * p.scale[n] + p.shift[n];
        if (p.act == 1)      c = fmaxf(c, 0.f);
        else if (p.act == 2) c = 0.5f * c * (1.f + erff(c * 0.70710678118f));
        else if (p.act == 3) c = 1.f / (1.f + __expf(-c));
        if (p.Res)
          c += p.Res[(long long)z * p.rB + (long long)m * p.rR +
                     (long long)n * p.rC];
        p.C[(long long)z * p.cB + (long long)m * p.cR + (long long)n * p.cC] = c;
      }
    }
}

// ---------------------------------------------------------------------------
// small helper kernels
// ---------------------------------------------------------------------------
__device__ __forceinline__ float block_sum(float v, float* sh) {
  sh[threadIdx.x] = v; __syncthreads();
  for (int off = 128; off > 0; off >>= 1) {
    if ((int)threadIdx.x < off) sh[threadIdx.x] += sh[threadIdx.x + off];
    __syncthreads();
  }
  float r = sh[0]; __syncthreads(); return r;
}
__device__ __forceinline__ float block_max(float v, float* sh) {
  sh[threadIdx.x] = v; __syncthreads();
  for (int off = 128; off > 0; off >>= 1) {
    if ((int)threadIdx.x < off)
      sh[threadIdx.x] = fmaxf(sh[threadIdx.x], sh[threadIdx.x + off]);
    __syncthreads();
  }
  float r = sh[0]; __syncthreads(); return r;
}

__global__ void scale_w_k(const float* w, const float* dw, float* o) {
  int idx = blockIdx.x * 256 + threadIdx.x;
  if (idx < 256 * 1024) o[idx] = w[idx] * dw[idx & 1023];
}

// depthwise 3x3 (dilated); input (B,256,S) CHW, output TRANSPOSED (B,S,256)
// so the following pointwise GEMM gets a k-contiguous A operand.
__global__ void dwconv_k(const float* in, const float* w, float* out, int d) {
  int idx = blockIdx.x * 256 + threadIdx.x;
  if (idx >= 32 * 256 * 576) return;
  int s = idx % 576; int t = idx / 576; int c = t & 255; int b = t >> 8;
  int x = s % 24, y = s / 24;
  const float* base = in + (long long)b * 589824 + (long long)c * 576;
  float acc = 0.f;
  for (int ky = 0; ky < 3; ++ky)
    for (int kx = 0; kx < 3; ++kx) {
      int yy = y + (ky - 1) * d, xx = x + (kx - 1) * d;
      if (yy >= 0 && yy < 24 && xx >= 0 && xx < 24)
        acc += w[c * 9 + ky * 3 + kx] * base[yy * 24 + xx];
    }
  out[(long long)b * 147456 + (long long)s * 256 + c] = acc;
}

__global__ void meanchan_k(const float* in, float* out, long long bS,
                           long long cS, int C, int S) {
  int blk = blockIdx.x; int c = blk % C; int b = blk / C;
  const float* base = in + (long long)b * bS + (long long)c * cS;
  float s = 0.f;
  for (int i = threadIdx.x; i < S; i += 256) s += base[i];
  __shared__ float sh[256];
  s = block_sum(s, sh);
  if (threadIdx.x == 0) out[blk] = s / (float)S;
}

__global__ void assemble_x_k(const float* cls, const float* ms, float* X) {
  long long idx = (long long)blockIdx.x * 256 + threadIdx.x;
  if (idx >= 32LL * 590848) return;
  int b = (int)(idx / 590848); int r = (int)(idx % 590848);
  X[idx] = (r < 1024) ? cls[b * 1024 + r] : ms[(long long)b * 589824 + (r - 1024)];
}

__global__ void ln_k(const float* src, float* dst, const float* g,
                     const float* bb, int width, float eps) {
  long long base = (long long)blockIdx.x * width;
  int tid = threadIdx.x;
  int per = (width + 255) >> 8;
  float v[8]; float s1 = 0.f, s2 = 0.f;
  for (int i = 0; i < per; ++i) {
    int idx = tid + i * 256;
    float x = (idx < width) ? src[base + idx] : 0.f;
    v[i] = x; s1 += x; s2 += x * x;
  }
  __shared__ float sh[256];
  s1 = block_sum(s1, sh);
  s2 = block_sum(s2, sh);
  float mean = s1 / width;
  float var  = s2 / width - mean * mean;
  float inv  = rsqrtf(var + eps);
  for (int i = 0; i < per; ++i) {
    int idx = tid + i * 256;
    if (idx < width) dst[base + idx] = (v[i] - mean) * inv * g[idx] + bb[idx];
  }
}

__global__ void attnprep_k(const float* qkv, float* E, float* L) {
  int row = blockIdx.x;
  long long qb = (long long)row * 3072;
  int tid = threadIdx.x;
  float qv[4], kv[4]; float sq = 0.f, sk = 0.f;
#pragma unroll
  for (int i = 0; i < 4; ++i) {
    int idx = tid + i * 256;
    float a = tanhf(qkv[qb + idx])        * 0.499f + 0.5f;
    float b = tanhf(qkv[qb + 1024 + idx]) * 0.499f + 0.5f;
    qv[i] = a; kv[i] = b; sq += a; sk += b;
  }
  __shared__ float sh[256];
  sq = block_sum(sq, sh);
  sk = block_sum(sk, sh);
  const float eps = 1e-8f;
  long long ob = (long long)row * 1024;
#pragma unroll
  for (int i = 0; i < 4; ++i) {
    int idx = tid + i * 256;
    float qp = fmaxf(qv[i] / (sq + eps), eps);
    float kp = fmaxf(kv[i] / (sk + eps), eps);
    E[ob + idx] = kp - qp;
    L[ob + idx] = logf(qp) - logf(kp);
  }
}

__global__ void softmax_row_k(float* x, int width, int stride) {
  long long base = (long long)blockIdx.x * stride;
  int tid = threadIdx.x;
  int per = (width + 255) >> 8;
  float v[4]; float mx = -3.4e38f;
  for (int i = 0; i < per; ++i) {
    int idx = tid + i * 256;
    float t = (idx < width) ? x[base + idx] : -3.4e38f;
    v[i] = t; mx = fmaxf(mx, t);
  }
  __shared__ float sh[256];
  mx = block_max(mx, sh);
  float s = 0.f;
  for (int i = 0; i < per; ++i) {
    int idx = tid + i * 256;
    if (idx < width) { v[i] = __expf(v[i] - mx); s += v[i]; }
  }
  s = block_sum(s, sh);
  float inv = 1.f / s;
  for (int i = 0; i < per; ++i) {
    int idx = tid + i * 256;
    if (idx < width) x[base + idx] = v[i] * inv;
  }
}

__global__ void diff_k(const float* qkv, float* af) {
  long long idx = (long long)blockIdx.x * 256 + threadIdx.x;
  if (idx >= 32LL * 590848) return;
  int b = (int)(idx / 590848); int r = (int)(idx % 590848);
  int t = r >> 10; int dd = r & 1023;
  float v = qkv[(long long)b * 1772544 + (long long)t * 3072 + 2048 + dd];
  af[idx] = v - af[idx];
}

__global__ void bnprep_k(const float* g1b, const float* g, const float* b,
                         const float* m, const float* v, float* sc, float* sf) {
  int n = blockIdx.x * 256 + threadIdx.x;
  if (n >= 2048) return;
  float inv = rsqrtf(v[n] + 1e-5f);
  float s = g[n] * inv;
  sc[n] = s;
  sf[n] = (g1b[n] - m[n]) * s + b[n];
}

// tiled transpose: in[b][c][s] (+inOff) -> out[b][s][c]
__global__ void transpose_k(const float* in, float* out, int C, int S,
                            long long inB, long long outB, int inOff) {
  __shared__ float t[32][33];
  int b = blockIdx.z;
  int s0 = blockIdx.x * 32, c0 = blockIdx.y * 32;
  int tx = threadIdx.x & 31, ty = threadIdx.x >> 5;  // 32x8
  const float* src = in + (long long)b * inB + inOff;
#pragma unroll
  for (int i = 0; i < 4; ++i) {
    int c = c0 + ty + i * 8, s = s0 + tx;
    t[ty + i * 8][tx] = (c < C && s < S) ? src[(long long)c * S + s] : 0.f;
  }
  __syncthreads();
  float* dst = out + (long long)b * outB;
#pragma unroll
  for (int i = 0; i < 4; ++i) {
    int s = s0 + ty + i * 8, c = c0 + tx;
    if (s < S && c < C) dst[(long long)s * C + c] = t[tx][ty + i * 8];
  }
}

// permute g2 weights [o][c][3][3] -> [o][q][c]  (k = q*2048 + c contiguous)
__global__ void wperm_k(const float* w, float* o) {
  int idx = blockIdx.x * 256 + threadIdx.x;
  if (idx >= 18874368) return;
  int n = idx / 18432; int r = idx - n * 18432;
  int q = r >> 11; int c = r & 2047;
  o[idx] = w[n * 18432 + c * 9 + q];
}

__global__ void gatefuse_k(const float* gate, const float* xs, float* gap) {
  int blk = blockIdx.x; int c = blk & 1023; int b = blk >> 10;
  const float* gb = gate + (long long)b * 589824 + (long long)c * 576;
  const float* rb = xs + (long long)b * 590848 + 1024 + (long long)c * 576;
  float s = 0.f;
  for (int i = threadIdx.x; i < 576; i += 256) s += gb[i] * rb[i];
  __shared__ float sh[256];
  s = block_sum(s, sh);
  if (threadIdx.x == 0) gap[blk] = s / 576.f;
}

__global__ void imgfeat_k(const float* xs, const float* gap, float* o) {
  int idx = blockIdx.x * 256 + threadIdx.x;
  if (idx >= 32 * 2048) return;
  int b = idx >> 11, j = idx & 2047;
  o[idx] = (j < 1024) ? xs[(long long)b * 590848 + j] : gap[b * 1024 + (j - 1024)];
}

__global__ void localfeat_k(const float* imf, const float* w, const float* bs,
                            float* out) {
  int idx = blockIdx.x * 256 + threadIdx.x;
  if (idx >= 32 * 2048) return;
  int b = idx >> 11, i = idx & 2047;
  const float* row = imf + (long long)b * 2048;
  float xm = (i > 0) ? row[i - 1] : 0.f;
  float x0 = row[i];
  float xp = (i < 2047) ? row[i + 1] : 0.f;
  out[idx] = w[0] * xm + w[1] * x0 + w[2] * xp + bs[0];
}

__global__ void fuse_k(const float* gf, const float* lf, const float* imf,
                       float* out) {
  int b = blockIdx.x; int tid = threadIdx.x;
  long long base = (long long)b * 2048;
  float gv[8], lv[8]; float gm = -3.4e38f, lm = -3.4e38f;
#pragma unroll
  for (int i = 0; i < 8; ++i) {
    int idx = tid + i * 256;
    gv[i] = gf[base + idx]; lv[i] = lf[base + idx];
    gm = fmaxf(gm, gv[i]); lm = fmaxf(lm, lv[i]);
  }
  __shared__ float sh[256];
  gm = block_max(gm, sh);
  lm = block_max(lm, sh);
  float gs = 0.f, ls = 0.f;
#pragma unroll
  for (int i = 0; i < 8; ++i) {
    gv[i] = __expf(gv[i] - gm); gs += gv[i];
    lv[i] = __expf(lv[i] - lm); ls += lv[i];
  }
  gs = block_sum(gs, sh);
  ls = block_sum(ls, sh);
#pragma unroll
  for (int i = 0; i < 8; ++i) {
    int idx = tid + i * 256;
    out[base + idx] = (gv[i] / gs + lv[i] / ls + 1.f) * imf[base + idx];
  }
}

// ---------------------------------------------------------------------------
// host orchestration
// ---------------------------------------------------------------------------
static void launch_gemm(hipStream_t st,
    const float* A, long long aB, long long aR, long long aC, int aMode,
    const float* Bw, long long bB, long long bR, long long bC, int bMode,
    float* C, long long cB, long long cR, long long cC,
    int M, int N, int K, int Z,
    const float* bias, const float* scale, const float* shift, int act,
    const float* Res, long long rB, long long rR, long long rC,
    float alpha, int Cc = 1, int Wc = 24, int Hc = 24) {
  GP p;
  p.A = A; p.Bw = Bw; p.C = C;
  p.bias = bias; p.scale = scale; p.shift = shift; p.Res = Res;
  p.aB = aB; p.aR = aR; p.aC = aC;
  p.bB = bB; p.bR = bR; p.bC = bC;
  p.cB = cB; p.cR = cR; p.cC = cC;
  p.rB = rB; p.rR = rR; p.rC = rC;
  p.M = M; p.N = N; p.K = K;
  p.aMode = aMode; p.bMode = bMode; p.Cc = Cc; p.Wc = Wc; p.Hc = Hc;
  p.act = act; p.alpha = alpha;
  dim3 grid((M + BMT - 1) / BMT, (N + BNT - 1) / BNT, Z);
  bool fA = (aC == 1) && ((aR & 3) == 0) && ((aB & 3) == 0);
  bool fB = (bMode == 0) && (bC == 1) && ((bR & 3) == 0) && ((bB & 3) == 0);
  if (fA && fB)      wmma_gemm<1, 1><<<grid, dim3(256), 0, st>>>(p);
  else if (fA)       wmma_gemm<1, 0><<<grid, dim3(256), 0, st>>>(p);
  else if (fB)       wmma_gemm<0, 1><<<grid, dim3(256), 0, st>>>(p);
  else               wmma_gemm<0, 0><<<grid, dim3(256), 0, st>>>(p);
}

// workspace layout (float offsets); aliased by lifetime. Needs ~551 MB.
static const size_t F_MS   = 0;           // ms (B,1024,576); later OUT, RT, CO
static const size_t F_HD   = 18874368;    // dwconv temp (B,S,256)
static const size_t F_QKV  = 23592960;    // qkv (B,577,3072); later HL; later W2P
static const size_t F_T1   = 42500096;    // mlp hidden (B,577,4096); later XS
static const size_t F_X    = 80314368;    // x tokens; later ATT; later GB (spans X+E)
static const size_t F_E    = 99221504;    // kp-qp; later attn_feat/diff
static const size_t F_L    = 118128640;   // log-ratio; later f_diff
static const size_t F_SM   = 137035776;   // small buffers
static const size_t F_WF   = F_SM + 0;
static const size_t F_GAP  = F_SM + 262144;
static const size_t F_HFC  = F_SM + 294912;
static const size_t F_CLS  = F_SM + 303104;
static const size_t F_BNS  = F_SM + 335872;
static const size_t F_BNT  = F_SM + 337920;
static const size_t F_IGAP = F_SM + 339968;
static const size_t F_IMF  = F_SM + 372736;
static const size_t F_FC1O = F_SM + 438272;
static const size_t F_GF   = F_SM + 471040;
static const size_t F_LF   = F_SM + 536576;
static const size_t F_FU   = F_SM + 602112;
static const size_t F_FOUT = F_SM + 667648;

extern "C" void kernel_launch(void* const* d_in, const int* in_sizes, int n_in,
                              void* d_out, int out_size, void* d_ws,
                              size_t ws_size, hipStream_t stream) {
  (void)in_sizes; (void)n_in; (void)out_size; (void)ws_size;
  const float* x         = (const float*)d_in[0];
  const float* sa_dw_w   = (const float*)d_in[1];
  const float* sa_pw_w   = (const float*)d_in[2];
  const float* sa_pw_b   = (const float*)d_in[3];
  const float* sa_fc1_w  = (const float*)d_in[4];
  const float* sa_fc1_b  = (const float*)d_in[5];
  const float* sa_fc2_w  = (const float*)d_in[6];
  const float* sa_fc2_b  = (const float*)d_in[7];
  const float* sa_ln_g   = (const float*)d_in[8];
  const float* sa_ln_b   = (const float*)d_in[9];
  const float* ad_ln_g   = (const float*)d_in[10];
  const float* ad_ln_b   = (const float*)d_in[11];
  const float* qkv_w     = (const float*)d_in[12];
  const float* qkv_b     = (const float*)d_in[13];
  const float* sp_diff_w = (const float*)d_in[14];
  const float* sp_diff_b = (const float*)d_in[15];
  const float* sp_ln_g   = (const float*)d_in[16];
  const float* sp_ln_b   = (const float*)d_in[17];
  const float* sp_mlp_w1 = (const float*)d_in[18];
  const float* sp_mlp_b1 = (const float*)d_in[19];
  const float* sp_mlp_w2 = (const float*)d_in[20];
  const float* sp_mlp_b2 = (const float*)d_in[21];
  const float* g1_w      = (const float*)d_in[22];
  const float* g1_b      = (const float*)d_in[23];
  const float* bn_g      = (const float*)d_in[24];
  const float* bn_b      = (const float*)d_in[25];
  const float* bn_m      = (const float*)d_in[26];
  const float* bn_v      = (const float*)d_in[27];
  const float* g2_w      = (const float*)d_in[28];
  const float* g2_b      = (const float*)d_in[29];
  const float* cn_g      = (const float*)d_in[30];
  const float* cn_b      = (const float*)d_in[31];
  const float* c1d_w     = (const float*)d_in[32];
  const float* c1d_b     = (const float*)d_in[33];
  const float* fc1_w     = (const float*)d_in[34];
  const float* fc1_b     = (const float*)d_in[35];
  const float* fc2_w     = (const float*)d_in[36];
  const float* fc2_b     = (const float*)d_in[37];
  const float* fcf_w     = (const float*)d_in[38];
  const float* fcf_b     = (const float*)d_in[39];
  const float* fln_g     = (const float*)d_in[40];
  const float* fln_b     = (const float*)d_in[41];
  const float* at_dw[3]  = {(const float*)d_in[42], (const float*)d_in[45],
                            (const float*)d_in[48]};
  const float* at_pw[3]  = {(const float*)d_in[43], (const float*)d_in[46],
                            (const float*)d_in[49]};
  const float* at_pb[3]  = {(const float*)d_in[44], (const float*)d_in[47],
                            (const float*)d_in[50]};
  float* W = (float*)d_ws;
  float* out = (float*)d_out;
  hipStream_t st = stream;

  // ---- spatial adapter -----------------------------------------------------
  scale_w_k<<<1024, 256, 0, st>>>(sa_pw_w, sa_dw_w, W + F_WF); // fold dw scale
  // b1 = pw(img*dw): tokens (skip cls) -> ms channels 0..255 (CHW)
  launch_gemm(st, x + 1024, 590848, 1024, 1, 0,
              W + F_WF, 0, 1024, 1, 0,
              W + F_MS, 589824, 1, 576,
              576, 256, 1024, 32, sa_pw_b, nullptr, nullptr, 0,
              nullptr, 0, 0, 0, 1.f);
  for (int d = 1; d <= 3; ++d) {
    dwconv_k<<<18432, 256, 0, st>>>(W + F_MS, at_dw[d - 1], W + F_HD, d);
    launch_gemm(st, W + F_HD, 147456, 256, 1, 0,        // (B,S,256) k-contig
                at_pw[d - 1], 0, 256, 1, 0,
                W + F_MS + (size_t)d * 147456, 589824, 1, 576,
                576, 256, 256, 32, at_pb[d - 1], nullptr, nullptr, 1,
                nullptr, 0, 0, 0, 1.f);
  }
  meanchan_k<<<32 * 1024, 256, 0, st>>>(W + F_MS, W + F_GAP, 589824, 576, 1024, 576);
  launch_gemm(st, W + F_GAP, 0, 1024, 1, 0, sa_fc1_w, 0, 1024, 1, 0,
              W + F_HFC, 0, 256, 1, 32, 256, 1024, 1,
              sa_fc1_b, nullptr, nullptr, 1, nullptr, 0, 0, 0, 1.f);
  launch_gemm(st, W + F_HFC, 0, 256, 1, 0, sa_fc2_w, 0, 256, 1, 0,
              W + F_CLS, 0, 1024, 1, 32, 1024, 256, 1,
              sa_fc2_b, nullptr, nullptr, 0, x, 0, 590848, 1, 1.f);
  assemble_x_k<<<73856, 256, 0, st>>>(W + F_CLS, W + F_MS, W + F_X);
  ln_k<<<18464, 256, 0, st>>>(W + F_X, W + F_X, sa_ln_g, sa_ln_b, 1024, 1e-5f);
  ln_k<<<18464, 256, 0, st>>>(W + F_X, W + F_X, ad_ln_g, ad_ln_b, 1024, 1e-5f);

  // ---- sinkhorn-ish attention ---------------------------------------------
  launch_gemm(st, W + F_X, 590848, 1024, 1, 0, qkv_w, 0, 1024, 1, 0,
              W + F_QKV, 1772544, 3072, 1, 577, 3072, 1024, 32,
              qkv_b, nullptr, nullptr, 0, nullptr, 0, 0, 0, 1.f);
  attnprep_k<<<18464, 256, 0, st>>>(W + F_QKV, W + F_E, W + F_L);
  // sid_sim: (B,577,577) padded row stride 584 for 16B-aligned rows
  launch_gemm(st, W + F_E, 590848, 1024, 1, 0, W + F_L, 590848, 1024, 1, 0,
              W + F_X, 336968, 584, 1, 577, 577, 1024, 32,
              nullptr, nullptr, nullptr, 0, nullptr, 0, 0, 0, 0.03125f);
  softmax_row_k<<<18464, 256, 0, st>>>(W + F_X, 577, 584);
  launch_gemm(st, W + F_X, 336968, 584, 1, 0,
              W + F_QKV + 2048, 1772544, 1, 3072, 0,
              W + F_E, 590848, 1024, 1, 577, 1024, 577, 32,
              nullptr, nullptr, nullptr, 0, nullptr, 0, 0, 0, 1.f);
  diff_k<<<73856, 256, 0, st>>>(W + F_QKV, W + F_E);
  launch_gemm(st, W + F_E, 590848, 1024, 1, 0, sp_diff_w, 0, 1024, 1, 0,
              W + F_L, 590848, 1024, 1, 577, 1024, 1024, 32,
              sp_diff_b, nullptr, nullptr, 0,
              W + F_QKV, 1772544, 3072, 1, 1.f);   // + q residual
  ln_k<<<18464, 256, 0, st>>>(W + F_L, W + F_QKV, sp_ln_g, sp_ln_b, 1024, 1e-5f);
  launch_gemm(st, W + F_QKV, 590848, 1024, 1, 0, sp_mlp_w1, 0, 1024, 1, 0,
              W + F_T1, 2363392, 4096, 1, 577, 4096, 1024, 32,
              sp_mlp_b1, nullptr, nullptr, 2, nullptr, 0, 0, 0, 1.f);
  launch_gemm(st, W + F_T1, 2363392, 4096, 1, 0, sp_mlp_w2, 0, 4096, 1, 0,
              W + F_MS, 590848, 1024, 1, 577, 1024, 4096, 32,
              sp_mlp_b2, nullptr, nullptr, 0, W + F_L, 590848, 1024, 1, 1.f);
  ln_k<<<18464, 256, 0, st>>>(W + F_MS, W + F_T1, sp_ln_g, sp_ln_b, 1024, 1e-5f);
  // W+F_T1 now holds xs (B,577,1024)

  // ---- gated conv path -----------------------------------------------------
  bnprep_k<<<8, 256, 0, st>>>(g1_b, bn_g, bn_b, bn_m, bn_v, W + F_BNS, W + F_BNT);
  // rgb_feat transposed to (B,S,1024) so g1's A operand is k-contiguous
  transpose_k<<<dim3(18, 32, 32), 256, 0, st>>>(W + F_T1, W + F_MS,
                                                1024, 576, 590848, 589824, 1024);
  launch_gemm(st, W + F_MS, 589824, 1024, 1, 0, g1_w, 0, 1024, 1, 0,
              W + F_X, 1179648, 2048, 1, 576, 2048, 1024, 32,   // GB (B,S,2048)
              nullptr, W + F_BNS, W + F_BNT, 1, nullptr, 0, 0, 0, 1.f);
  // permute g2 weights to k-major [o][q][c] so conv B uses the vector path
  wperm_k<<<73728, 256, 0, st>>>(g2_w, W + F_QKV);
  // implicit-im2col 3x3 conv 2048->1024 + bias + sigmoid gate, single launch
  launch_gemm(st, W + F_X, 1179648, 2048, 1, 1,
              W + F_QKV, 0, 18432, 1, 0,
              W + F_MS, 589824, 1, 576, 576, 1024, 18432, 32,
              g2_b, nullptr, nullptr, 3, nullptr, 0, 0, 0, 1.f, 2048, 24, 24);
  gatefuse_k<<<32 * 1024, 256, 0, st>>>(W + F_MS, W + F_T1, W + F_IGAP);

  // ---- head ----------------------------------------------------------------
  imgfeat_k<<<256, 256, 0, st>>>(W + F_T1, W + F_IGAP, W + F_IMF);
  ln_k<<<32, 256, 0, st>>>(W + F_IMF, W + F_IMF, cn_g, cn_b, 2048, 1e-5f);
  launch_gemm(st, W + F_IMF, 0, 2048, 1, 0, fc1_w, 0, 2048, 1, 0,
              W + F_FC1O, 0, 1024, 1, 32, 1024, 2048, 1,
              fc1_b, nullptr, nullptr, 0, nullptr, 0, 0, 0, 1.f);
  launch_gemm(st, W + F_FC1O, 0, 1024, 1, 0, fc2_w, 0, 1024, 1, 0,
              W + F_GF, 0, 2048, 1, 32, 2048, 1024, 1,
              fc2_b, nullptr, nullptr, 0, nullptr, 0, 0, 0, 1.f);
  localfeat_k<<<256, 256, 0, st>>>(W + F_IMF, c1d_w, c1d_b, W + F_LF);
  fuse_k<<<32, 256, 0, st>>>(W + F_GF, W + F_LF, W + F_IMF, W + F_FU);
  launch_gemm(st, W + F_FU, 0, 2048, 1, 0, fcf_w, 0, 2048, 1, 0,
              W + F_FOUT, 0, 1024, 1, 32, 1024, 2048, 1,
              fcf_b, nullptr, nullptr, 0, nullptr, 0, 0, 0, 1.f);
  ln_k<<<32, 256, 0, st>>>(W + F_FOUT, out, fln_g, fln_b, 1024, 1e-5f);
}
